// SandwichRotationGeoDynamicLayer_37048387895889
// MI455X (gfx1250) — matrix operand order repaired and checked by reading, not compiled, not run on hardware
//
#include <hip/hip_runtime.h>
#include <math.h>

// ---------------------------------------------------------------------------
// Problem constants
// ---------------------------------------------------------------------------
#define BATCH   32
#define NSEQ    512
#define INF     1024
#define OUTF    4096
#define HID     128
#define KROT    8
#define RANK    8
#define R2      128          // 2*KROT*RANK : low-rank width
#define MTOT    16384        // BATCH*NSEQ
#define KTOT    1152         // INF + R2
#define ROT_GATE_BIAS   (-4.0f)
#define SCALE_GATE_BIAS (-2.5f)
#define RESIDUAL_SCALE  1.0f
#define ADAPTER_SCALE   1.0f

// ---------------------------------------------------------------------------
// Types for WMMA (CDNA5 / gfx1250, wave32)
// ---------------------------------------------------------------------------
typedef __attribute__((ext_vector_type(16))) __bf16 bf16x16;
typedef __attribute__((ext_vector_type(8)))  __bf16 bf16x8;
typedef __attribute__((ext_vector_type(4)))  __bf16 bf16x4;
typedef __attribute__((ext_vector_type(8)))  float  f32x8;
typedef __attribute__((ext_vector_type(4)))  int    i32x4;

union FragU { bf16x16 v; bf16x8 h[2]; };

__device__ __forceinline__ f32x8 wmma_bf16(bf16x16 a, bf16x16 b, f32x8 c) {
  return __builtin_amdgcn_wmma_f32_16x16x32_bf16(false, a, false, b,
                                                 (short)0, c, false, false);
}

__device__ __forceinline__ float sigmoidf_(float x) { return 1.0f / (1.0f + expf(-x)); }
__device__ __forceinline__ float geluf_(float x) {
  float x3 = x * x * x;
  return 0.5f * x * (1.0f + tanhf(0.7978845608028654f * (x + 0.044715f * x3)));
}

// ---------------------------------------------------------------------------
// Async global->LDS copy (CDNA5 ASYNCcnt path) with safe fallback
// ---------------------------------------------------------------------------
#if defined(__gfx1250__) && \
    __has_builtin(__builtin_amdgcn_global_load_async_to_lds_b128) && \
    __has_builtin(__builtin_amdgcn_s_wait_asynccnt)
#define HAVE_ASYNC_LDS 1
typedef __attribute__((address_space(1))) i32x4 gv_i32x4;   // global int4
typedef __attribute__((address_space(3))) i32x4 lv_i32x4;   // LDS int4
#endif

// copies nfloat (multiple of 4) contiguous floats; caller must __syncthreads()
__device__ __forceinline__ void block_copy_f32(const float* __restrict__ gsrc,
                                               float* __restrict__ lds,
                                               int nfloat, int tid, int nthreads) {
#ifdef HAVE_ASYNC_LDS
  int nchunk = nfloat >> 2;
  for (int c = tid; c < nchunk; c += nthreads) {
    __builtin_amdgcn_global_load_async_to_lds_b128(
        (gv_i32x4*)(const_cast<float*>(gsrc) + (c << 2)),
        (lv_i32x4*)(lds + (c << 2)), 0, 0);
  }
  __builtin_amdgcn_s_wait_asynccnt(0);
#else
  for (int c = tid; c < nfloat; c += nthreads) lds[c] = gsrc[c];
#endif
}

// ---------------------------------------------------------------------------
// K1: per-(batch,feature) stats -> z = [cls | mean | var]  (32 x 3072)
// ---------------------------------------------------------------------------
__global__ __launch_bounds__(256) void stats_kernel(const float* __restrict__ X,
                                                    float* __restrict__ Z) {
  int g = blockIdx.x * 256 + threadIdx.x;          // 32768 = 32*1024
  int b = g >> 10, d = g & 1023;
  const float* xb = X + (size_t)b * NSEQ * INF + d;
  float cls = xb[0];
  float s = 0.f, s2 = 0.f;
  for (int n = 0; n < NSEQ; ++n) {
    float v = xb[(size_t)n * INF];
    s += v; s2 += v * v;
  }
  float mean = s * (1.0f / NSEQ);
  float var  = s2 * (1.0f / NSEQ) - mean * mean;
  Z[b * 3072 + d]        = cls;
  Z[b * 3072 + 1024 + d] = mean;
  Z[b * 3072 + 2048 + d] = var;
}

// ---------------------------------------------------------------------------
// K2: build Ut = [L|R]^T (128x1024) and Wt = [R|-L]^T (128x1024)
// left/right layout: [k][d][r] = k*8192 + d*8 + r
// ---------------------------------------------------------------------------
__global__ __launch_bounds__(256) void prep_kernel(const float* __restrict__ left,
                                                   const float* __restrict__ right,
                                                   float* __restrict__ Ut,
                                                   float* __restrict__ Wt) {
  int g = blockIdx.x * 256 + threadIdx.x;          // 131072 = 128*1024
  int j = g >> 10, d = g & 1023;
  if (j < 64) {
    int k = j >> 3, r = j & 7;
    Ut[g] = left [k * 8192 + d * 8 + r];
    Wt[g] = right[k * 8192 + d * 8 + r];
  } else {
    int k = (j - 64) >> 3, r = (j - 64) & 7;
    Ut[g] = right[k * 8192 + d * 8 + r];
    Wt[g] = -left[k * 8192 + d * 8 + r];
  }
}

// ---------------------------------------------------------------------------
// K3: control head (single block): Gram norms, h = gelu(z W1^T + b1),
//     rot/scale coeffs, gates.  a[b][k] = RESIDUAL*rot_coeff/norm_k.
// ---------------------------------------------------------------------------
__global__ __launch_bounds__(256) void ctrl_kernel(
    const float* __restrict__ Z, const float* __restrict__ left,
    const float* __restrict__ right,
    const float* __restrict__ cw1, const float* __restrict__ cb1,
    const float* __restrict__ rcw, const float* __restrict__ rcb,
    const float* __restrict__ rgw, const float* __restrict__ rgb,
    const float* __restrict__ scw, const float* __restrict__ scb,
    const float* __restrict__ sgw, const float* __restrict__ sgb,
    float* __restrict__ Acoef, float* __restrict__ Gamma,
    float* __restrict__ Sgate, float* __restrict__ Sccoef) {
  __shared__ float gram[KROT][3][64];   // [k][{L^TL, R^TR, R^TL}][r1*8+r2]
  __shared__ float nrm[KROT];
  __shared__ float hsh[BATCH][HID];
  int tid = threadIdx.x;

  // --- Gram matrices (8x8 each) ---
  for (int t = tid; t < KROT * 3 * 64; t += 256) {
    int k = t / 192; int rem = t % 192; int m = rem / 64; int idx = rem % 64;
    int r1 = idx >> 3, r2 = idx & 7;
    const float* P1 = (m == 0) ? left : right;          // m0:L^TL  m1:R^TR  m2:R^TL
    const float* P2 = (m == 1) ? right : left;
    const float* p1 = P1 + k * 8192 + r1;
    const float* p2 = P2 + k * 8192 + r2;
    float s = 0.f;
    for (int d = 0; d < INF; ++d) s += p1[d * 8] * p2[d * 8];
    gram[k][m][idx] = s;
  }
  __syncthreads();
  if (tid < KROT) {
    int k = tid; float tA = 0.f, tB = 0.f;
    for (int i = 0; i < 8; ++i)
      for (int j = 0; j < 8; ++j) {
        tA += gram[k][0][i * 8 + j] * gram[k][1][i * 8 + j];   // tr(Gl Gr)
        tB += gram[k][2][i * 8 + j] * gram[k][2][j * 8 + i];   // tr(C^2)
      }
    float n2 = 2.0f * (tA - tB);
    nrm[k] = fmaxf(sqrtf(fmaxf(n2, 0.f)), 1e-6f);
  }
  // --- hidden layer h ---
  __syncthreads();
  for (int t = tid; t < BATCH * HID; t += 256) {
    int b = t >> 7, i = t & 127;
    const float* zr = Z + b * 3072;
    const float* wr = cw1 + i * 3072;
    float s = cb1[i];
    for (int d = 0; d < 3072; ++d) s += zr[d] * wr[d];
    hsh[b][i] = geluf_(s);
  }
  __syncthreads();
  // --- coefficient heads (256 items) ---
  {
    int b = tid >> 3, k = tid & 7;
    float s = rcb[k], s2 = scb[k];
    for (int i = 0; i < HID; ++i) {
      float hv = hsh[b][i];
      s  += hv * rcw[k * HID + i];
      s2 += hv * scw[k * HID + i];
    }
    Acoef[b * 8 + k]  = RESIDUAL_SCALE * s / nrm[k];
    Sccoef[b * 8 + k] = s2;
  }
  // --- gates ---
  if (tid < BATCH) {
    int b = tid;
    float s = rgb[0] + ROT_GATE_BIAS;
    float s2 = sgb[0] + SCALE_GATE_BIAS;
    for (int i = 0; i < HID; ++i) {
      float hv = hsh[b][i];
      s += hv * rgw[i]; s2 += hv * sgw[i];
    }
    Gamma[b] = ADAPTER_SCALE * sigmoidf_(s);
    Sgate[b] = ADAPTER_SCALE * sigmoidf_(s2);
  }
}

// ---------------------------------------------------------------------------
// K4: G = Wt * Ut^T  (128x128, dot over 1024)
// ---------------------------------------------------------------------------
__global__ __launch_bounds__(256) void g_kernel(const float* __restrict__ Ut,
                                                const float* __restrict__ Wt,
                                                float* __restrict__ G) {
  int g = blockIdx.x * 256 + threadIdx.x;          // 16384
  int i = g >> 7, j = g & 127;
  const float* wi = Wt + i * 1024;
  const float* uj = Ut + j * 1024;
  float s = 0.f;
  for (int d = 0; d < 1024; ++d) s += wi[d] * uj[d];
  G[g] = s;
}

// ---------------------------------------------------------------------------
// K5: P = Ws * W   (4096 x 128).  grid (16 o-blocks, 8 j-blocks)
// ---------------------------------------------------------------------------
__global__ __launch_bounds__(256) void p_kernel(const float* __restrict__ Ws,
                                                const float* __restrict__ Wt,
                                                float* __restrict__ P) {
  __shared__ float wtile[16][1024];                // 64 KB
  int bo = blockIdx.x, bj = blockIdx.y, tid = threadIdx.x;
  block_copy_f32(Wt + (size_t)(bj * 16) * 1024, &wtile[0][0], 16 * 1024, tid, 256);
  __syncthreads();
  int o = bo * 256 + tid;
  const float* wr = Ws + (size_t)o * 1024;
  float acc[16];
#pragma unroll
  for (int jj = 0; jj < 16; ++jj) acc[jj] = 0.f;
  for (int d = 0; d < 1024; ++d) {
    float w = wr[d];
#pragma unroll
    for (int jj = 0; jj < 16; ++jj) acc[jj] += w * wtile[jj][d];
  }
#pragma unroll
  for (int jj = 0; jj < 16; ++jj) P[(size_t)o * 128 + bj * 16 + jj] = acc[jj];
}

// ---------------------------------------------------------------------------
// K6: per-batch  M_b = I - 0.5 * D_b * G ;  invert (Gauss-Jordan, no pivot,
//     M ~= I so it is safe) ;  E_b = gamma_b * M^-1 * D_b
// ---------------------------------------------------------------------------
__global__ __launch_bounds__(128) void e_kernel(const float* __restrict__ G,
                                                const float* __restrict__ Acoef,
                                                const float* __restrict__ Gamma,
                                                float* __restrict__ E) {
  __shared__ float M[128][129];                    // padded vs bank conflicts
  int b = blockIdx.x, i = threadIdx.x;
  float di = Acoef[b * 8 + ((i & 63) >> 3)];
  for (int j = 0; j < 128; ++j)
    M[i][j] = ((i == j) ? 1.0f : 0.0f) - 0.5f * di * G[i * 128 + j];
  __syncthreads();
  for (int p = 0; p < 128; ++p) {
    float f   = M[i][p];       // own row: safe
    float rpp = M[p][p];       // row p untouched since last barrier
    __syncthreads();
    float ipiv = 1.0f / rpp;
    if (i == p) {
      for (int j = 0; j < 128; ++j) M[p][j] = (j == p) ? ipiv : M[p][j] * ipiv;
    }
    __syncthreads();
    if (i != p) {
      for (int j = 0; j < 128; ++j)
        if (j != p) M[i][j] -= f * M[p][j];
      M[i][p] = -f * ipiv;
    }
    __syncthreads();
  }
  float g = Gamma[b];
  for (int j = 0; j < 128; ++j) {
    float dj = Acoef[b * 8 + ((j & 63) >> 3)];
    E[(size_t)b * 16384 + i * 128 + j] = g * M[i][j] * dj;
  }
}

// ---------------------------------------------------------------------------
// WMMA-GEMM machinery: f32 tiles staged to LDS as bf16 hi/lo pairs.
// Block tile 128x128x32, 8 waves, each wave: 4x2 accumulators of 16x16.
// Double-buffered LDS (one barrier per k-step), prefetch at distance 2.
// ---------------------------------------------------------------------------
#define LDA_T 40   // padded LDS row stride (bf16 elements)

__device__ __forceinline__ void stage_tile(const float* __restrict__ src, int ld,
                                           __bf16* dh, __bf16* dl, int tid) {
  // 128 rows x 32 cols f32 -> hi/lo bf16, 4 float4 chunks per thread
#pragma unroll
  for (int it = 0; it < 4; ++it) {
    int c   = tid + it * 256;
    int row = c >> 3;
    int col = (c & 7) << 2;
    float4 v = *(const float4*)(src + (size_t)row * ld + col);
    __bf16 h0 = (__bf16)v.x, h1 = (__bf16)v.y, h2 = (__bf16)v.z, h3 = (__bf16)v.w;
    bf16x4 hv = {h0, h1, h2, h3};
    bf16x4 lv = {(__bf16)(v.x - (float)h0), (__bf16)(v.y - (float)h1),
                 (__bf16)(v.z - (float)h2), (__bf16)(v.w - (float)h3)};
    *(bf16x4*)(dh + row * LDA_T + col) = hv;
    *(bf16x4*)(dl + row * LDA_T + col) = lv;
  }
}

__device__ __forceinline__ void prefetch_tile(const float* src, int ld, int tid) {
  // one byte per thread touches each 64B half-row of the 128x32 f32 tile
  __builtin_prefetch(src + (size_t)(tid >> 1) * ld + (tid & 1) * 16, 0, 1);
}

__device__ __forceinline__ FragU load_frag_a(const __bf16* s, int mt, int lane) {
  // A frag (16x32 bf16): M = lane%16, elem e<8: K=8h+e ; e>=8: K=16+8h+(e-8)
  int hf = lane >> 4;
  const __bf16* p = s + (mt + (lane & 15)) * LDA_T + 8 * hf;
  FragU u;
  u.h[0] = *(const bf16x8*)p;
  u.h[1] = *(const bf16x8*)(p + 16);
  return u;
}

__device__ __forceinline__ FragU load_frag_b(const __bf16* s, int nt, int lane) {
  // B frag (32x16 bf16): N = lane%16, elem e<8: K=8h+e ; e>=8: K=16+8h+(e-8)  (B stored [N][K])
  int hf = lane >> 4;
  const __bf16* p = s + (nt + (lane & 15)) * LDA_T + 8 * hf;
  FragU u;
  u.h[0] = *(const bf16x8*)p;
  u.h[1] = *(const bf16x8*)(p + 16);
  return u;
}

// ---------------------------------------------------------------------------
// K7: t1 = x @ U   (16384x1024 * 1024x128), bf16x3 WMMA, grid 128 blocks
// ---------------------------------------------------------------------------
__global__ __launch_bounds__(256) void t1_kernel(const float* __restrict__ X,
                                                 const float* __restrict__ Ut,
                                                 float* __restrict__ T1) {
  __shared__ __bf16 sAh[2][128 * LDA_T], sAl[2][128 * LDA_T];
  __shared__ __bf16 sBh[2][128 * LDA_T], sBl[2][128 * LDA_T];
  int tid = threadIdx.x, bm = blockIdx.x;
  int wid = tid >> 5, lane = tid & 31;
  int wr = wid >> 2, wc = wid & 3;
  const float* Arow = X + (size_t)(bm * 128) * 1024;
  f32x8 acc[4][2];
  f32x8 zero = {0.f, 0.f, 0.f, 0.f, 0.f, 0.f, 0.f, 0.f};
#pragma unroll
  for (int i = 0; i < 4; ++i)
#pragma unroll
    for (int j = 0; j < 2; ++j) acc[i][j] = zero;

  stage_tile(Arow, 1024, sAh[0], sAl[0], tid);
  stage_tile(Ut,   1024, sBh[0], sBl[0], tid);
  __syncthreads();

  for (int kt = 0; kt < 32; ++kt) {
    int cur = kt & 1;
    if (kt + 1 < 32) {
      int kb = (kt + 1) * 32;
      stage_tile(Arow + kb, 1024, sAh[cur ^ 1], sAl[cur ^ 1], tid);
      stage_tile(Ut + kb,   1024, sBh[cur ^ 1], sBl[cur ^ 1], tid);
      if (kt + 2 < 32) {
        int kp = (kt + 2) * 32;
        prefetch_tile(Arow + kp, 1024, tid);
        prefetch_tile(Ut + kp,   1024, tid);
      }
    }
    FragU Ah[4], Al[4], Bh[2], Bl[2];
#pragma unroll
    for (int i = 0; i < 4; ++i) {
      Ah[i] = load_frag_a(sAh[cur], wr * 64 + i * 16, lane);
      Al[i] = load_frag_a(sAl[cur], wr * 64 + i * 16, lane);
    }
#pragma unroll
    for (int j = 0; j < 2; ++j) {
      Bh[j] = load_frag_b(sBh[cur], wc * 32 + j * 16, lane);
      Bl[j] = load_frag_b(sBl[cur], wc * 32 + j * 16, lane);
    }
#pragma unroll
    for (int i = 0; i < 4; ++i)
#pragma unroll
      for (int j = 0; j < 2; ++j) {
        acc[i][j] = wmma_bf16(Ah[i].v, Bh[j].v, acc[i][j]);
        acc[i][j] = wmma_bf16(Ah[i].v, Bl[j].v, acc[i][j]);
        acc[i][j] = wmma_bf16(Al[i].v, Bh[j].v, acc[i][j]);
      }
    __syncthreads();
  }
#pragma unroll
  for (int i = 0; i < 4; ++i)
#pragma unroll
    for (int j = 0; j < 2; ++j) {
      int row0 = bm * 128 + wr * 64 + i * 16 + ((lane >> 4) << 3);
      int col  = wc * 32 + j * 16 + (lane & 15);
#pragma unroll
      for (int r = 0; r < 8; ++r)
        T1[(size_t)(row0 + r) * 128 + col] = acc[i][j][r];
    }
}

// ---------------------------------------------------------------------------
// K8: t3 = t1 @ E_b  (per 128-row chunk; E_b staged to LDS, async path)
// ---------------------------------------------------------------------------
__global__ __launch_bounds__(256) void t3_kernel(const float* __restrict__ T1,
                                                 const float* __restrict__ E,
                                                 float* __restrict__ T3) {
  __shared__ float Es[128][128];                   // 64 KB
  int blk = blockIdx.x;                            // 128 chunks of 128 rows
  int b = blk >> 2;
  int tid = threadIdx.x;
  block_copy_f32(E + (size_t)b * 16384, &Es[0][0], 16384, tid, 256);
  __syncthreads();
  int row = blk * 128 + (tid >> 1);
  int hf  = tid & 1;
  const float* t1r = T1 + (size_t)row * 128;
  float acc[64];
#pragma unroll
  for (int j = 0; j < 64; ++j) acc[j] = 0.f;
  for (int m = 0; m < 128; ++m) {
    float f = t1r[m];
#pragma unroll
    for (int j = 0; j < 64; ++j) acc[j] += f * Es[m][hf * 64 + j];
  }
#pragma unroll
  for (int j = 0; j < 64; ++j)
    T3[(size_t)row * 128 + hf * 64 + j] = acc[j];
}

// ---------------------------------------------------------------------------
// K9: per-(b,o) output scale table
// ---------------------------------------------------------------------------
__global__ __launch_bounds__(256) void sc_kernel(const float* __restrict__ Sccoef,
                                                 const float* __restrict__ Sgate,
                                                 const float* __restrict__ sbasis,
                                                 float* __restrict__ SC) {
  int g = blockIdx.x * 256 + threadIdx.x;          // 131072 = 32*4096
  int b = g >> 12, o = g & 4095;
  float s = 0.f;
#pragma unroll
  for (int k = 0; k < 8; ++k) s += Sccoef[b * 8 + k] * sbasis[k * 4096 + o];
  SC[g] = 1.0f + Sgate[b] * tanhf(s);
}

// ---------------------------------------------------------------------------
// K10: main GEMM  out = ([x|t3] @ [Ws|P]^T + bias) * sc   (K = 1152)
// ---------------------------------------------------------------------------
__device__ __forceinline__ void main_srcs(int kt, int bm, int bn,
    const float* X, const float* T3, const float* Ws, const float* P,
    const float*& a, const float*& b2, int& la, int& lb) {
  int kb = kt * 32;
  if (kb < 1024) {
    a  = X  + (size_t)(bm * 128) * 1024 + kb; la = 1024;
    b2 = Ws + (size_t)(bn * 128) * 1024 + kb; lb = 1024;
  } else {
    int k2 = kb - 1024;
    a  = T3 + (size_t)(bm * 128) * 128 + k2;  la = 128;
    b2 = P  + (size_t)(bn * 128) * 128 + k2;  lb = 128;
  }
}

__global__ __launch_bounds__(256) void main_gemm_kernel(
    const float* __restrict__ X,  const float* __restrict__ T3,
    const float* __restrict__ Ws, const float* __restrict__ P,
    const float* __restrict__ bias, const float* __restrict__ SC,
    float* __restrict__ Out) {
  __shared__ __bf16 sAh[2][128 * LDA_T], sAl[2][128 * LDA_T];
  __shared__ __bf16 sBh[2][128 * LDA_T], sBl[2][128 * LDA_T];
  int tid = threadIdx.x;
  int bm = blockIdx.x, bn = blockIdx.y;
  int wid = tid >> 5, lane = tid & 31;
  int wr = wid >> 2, wc = wid & 3;
  f32x8 acc[4][2];
  f32x8 zero = {0.f, 0.f, 0.f, 0.f, 0.f, 0.f, 0.f, 0.f};
#pragma unroll
  for (int i = 0; i < 4; ++i)
#pragma unroll
    for (int j = 0; j < 2; ++j) acc[i][j] = zero;

  {
    const float *a0, *b0; int la, lb;
    main_srcs(0, bm, bn, X, T3, Ws, P, a0, b0, la, lb);
    stage_tile(a0, la, sAh[0], sAl[0], tid);
    stage_tile(b0, lb, sBh[0], sBl[0], tid);
  }
  __syncthreads();

  for (int kt = 0; kt < 36; ++kt) {
    int cur = kt & 1;
    if (kt + 1 < 36) {
      const float *a1, *b1; int la, lb;
      main_srcs(kt + 1, bm, bn, X, T3, Ws, P, a1, b1, la, lb);
      stage_tile(a1, la, sAh[cur ^ 1], sAl[cur ^ 1], tid);
      stage_tile(b1, lb, sBh[cur ^ 1], sBl[cur ^ 1], tid);
      if (kt + 2 < 36) {
        const float *a2, *b2; int la2, lb2;
        main_srcs(kt + 2, bm, bn, X, T3, Ws, P, a2, b2, la2, lb2);
        prefetch_tile(a2, la2, tid);
        prefetch_tile(b2, lb2, tid);
      }
    }
    FragU Ah[4], Al[4], Bh[2], Bl[2];
#pragma unroll
    for (int i = 0; i < 4; ++i) {
      Ah[i] = load_frag_a(sAh[cur], wr * 64 + i * 16, lane);
      Al[i] = load_frag_a(sAl[cur], wr * 64 + i * 16, lane);
    }
#pragma unroll
    for (int j = 0; j < 2; ++j) {
      Bh[j] = load_frag_b(sBh[cur], wc * 32 + j * 16, lane);
      Bl[j] = load_frag_b(sBl[cur], wc * 32 + j * 16, lane);
    }
#pragma unroll
    for (int i = 0; i < 4; ++i)
#pragma unroll
      for (int j = 0; j < 2; ++j) {
        acc[i][j] = wmma_bf16(Ah[i].v, Bh[j].v, acc[i][j]);
        acc[i][j] = wmma_bf16(Ah[i].v, Bl[j].v, acc[i][j]);
        acc[i][j] = wmma_bf16(Al[i].v, Bh[j].v, acc[i][j]);
      }
    __syncthreads();
  }
  int b = bm >> 2;                                 // 128-row tiles never cross a batch
#pragma unroll
  for (int i = 0; i < 4; ++i)
#pragma unroll
    for (int j = 0; j < 2; ++j) {
      int row0 = bm * 128 + wr * 64 + i * 16 + ((lane >> 4) << 3);
      int col  = bn * 128 + wc * 32 + j * 16 + (lane & 15);
      float sv = SC[b * 4096 + col];
      float bv = bias[col];
#pragma unroll
      for (int r = 0; r < 8; ++r)
        Out[(size_t)(row0 + r) * 4096 + col] = (acc[i][j][r] + bv) * sv;
    }
}

// ---------------------------------------------------------------------------
// Host launcher
// ---------------------------------------------------------------------------
extern "C" void kernel_launch(void* const* d_in, const int* in_sizes, int n_in,
                              void* d_out, int out_size, void* d_ws, size_t ws_size,
                              hipStream_t stream) {
  const float* X      = (const float*)d_in[0];
  const float* Ws     = (const float*)d_in[1];
  const float* bias   = (const float*)d_in[2];
  const float* left   = (const float*)d_in[3];
  const float* right  = (const float*)d_in[4];
  const float* sbasis = (const float*)d_in[5];
  const float* cw1    = (const float*)d_in[6];
  const float* cb1    = (const float*)d_in[7];
  const float* rcw    = (const float*)d_in[8];
  const float* rcb    = (const float*)d_in[9];
  const float* rgw    = (const float*)d_in[10];
  const float* rgb    = (const float*)d_in[11];
  const float* scw    = (const float*)d_in[12];
  const float* scb    = (const float*)d_in[13];
  const float* sgw    = (const float*)d_in[14];
  const float* sgb    = (const float*)d_in[15];
  float* Out = (float*)d_out;

  float* wsf = (float*)d_ws;
  // workspace layout (float offsets)
  float* Z      = wsf + 0;          //  98304
  float* Acoef  = wsf + 98304;      //    256
  float* Gamma  = wsf + 98560;      //     32
  float* Sgate  = wsf + 98592;      //     32
  float* Sccoef = wsf + 98624;      //    256
  float* Ut     = wsf + 98944;      // 131072
  float* Wt     = wsf + 230016;     // 131072
  float* G      = wsf + 361088;     //  16384
  float* P      = wsf + 377472;     // 524288
  float* E      = wsf + 901760;     // 524288
  float* SC     = wsf + 1426048;    // 131072
  float* T1     = wsf + 1557120;    // 2097152
  float* T3     = wsf + 3654272;    // 2097152  -> total ~23 MB

  stats_kernel<<<128, 256, 0, stream>>>(X, Z);
  prep_kernel <<<512, 256, 0, stream>>>(left, right, Ut, Wt);
  ctrl_kernel <<<1,   256, 0, stream>>>(Z, left, right, cw1, cb1, rcw, rcb,
                                        rgw, rgb, scw, scb, sgw, sgb,
                                        Acoef, Gamma, Sgate, Sccoef);
  g_kernel    <<<64,  256, 0, stream>>>(Ut, Wt, G);
  p_kernel    <<<dim3(16, 8), 256, 0, stream>>>(Ws, Wt, P);
  e_kernel    <<<32,  128, 0, stream>>>(G, Acoef, Gamma, E);
  t1_kernel   <<<128, 256, 0, stream>>>(X, Ut, T1);
  t3_kernel   <<<128, 256, 0, stream>>>(T1, E, T3);
  sc_kernel   <<<512, 256, 0, stream>>>(Sccoef, Sgate, sbasis, SC);
  main_gemm_kernel<<<dim3(128, 32), 256, 0, stream>>>(X, T3, Ws, P, bias, SC, Out);
}
